// GeneratorEncDec_18571438588679
// MI455X (gfx1250) — matrix operand order repaired and checked by reading, not compiled, not run on hardware
//
#include <hip/hip_runtime.h>
#include <hip/hip_bf16.h>
#include <math.h>

// Problem constants (match reference)
#define B_ 32
#define S_ 64
#define T_ 128
#define V_ 32000
#define E_ 512
#define H_ 1024
#define G_ 3072 /* 3*H */

typedef __attribute__((ext_vector_type(16))) _Float16 v16h;
typedef __attribute__((ext_vector_type(8)))  float    v8f;

#if defined(__AMDGCN__) && __has_builtin(__builtin_amdgcn_global_load_async_to_lds_b128)
#define USE_ASYNC_LDS 1
#else
#define USE_ASYNC_LDS 0
#endif

struct GemmJob {
  const _Float16* A;    // [M,K] f16 row-major (produced in f16 by upstream kernels)
  const _Float16* W;    // [N,K] f16 row-major (C = A * W^T)
  const float*    bias; // [N]
  float*          C;    // [M,ldc] f32
};

struct DirPtrs {
  const float* xg; // [B,S,3H] precomputed input gates (f32)
  const float* gh; // [B,3H]   hidden gates this step (f32)
  float*       h;  // [B,H]    hidden state f32 (in/out)
  _Float16*    hf; // [B,H]    hidden state f16 mirror (GEMM A operand)
};

// ---- WMMA fragment helpers (C/D per CDNA5 ISA 7.12.2; A/B in contiguous k order) --------------------

// A-matrix 16x32 f16: lane<16 holds row m0+lane, K = k0+e (e<16);
// lanes>=16 shift K by +16. Two b128 loads per lane.
__device__ __forceinline__ v16h load_a_frag(const _Float16* A, int lda, int m0, int k0, int lane) {
  const int m  = m0 + (lane & 15);
  const int hi = lane >> 4;
  const _Float16* p = A + (size_t)m * lda + (k0 + (hi << 4));
  v16h a;
#pragma unroll
  for (int e = 0; e < 8; ++e) a[e] = p[e];
#pragma unroll
  for (int e = 0; e < 8; ++e) a[e + 8] = p[e + 8];
  return a;
}

// B-matrix 32x16 f16 (B[k][n] = W[n][k]): lane&15 selects column n,
// lanes 0-15 cover K=k0..k0+15, lanes 16-31 cover K=k0+16..k0+31, e sequential.
__device__ __forceinline__ v16h load_b_frag_g(const _Float16* W, int ldw, int n0, int k0, int lane) {
  const int n  = n0 + (lane & 15);
  const int kb = k0 + ((lane >> 4) << 4);
  const _Float16* p = W + (size_t)n * ldw + kb;
  v16h b;
#pragma unroll
  for (int e = 0; e < 16; ++e) b[e] = p[e];
  return b;
}

// Same B fragment, read from an LDS tile laid out as [64 rows][32 K] f16.
__device__ __forceinline__ v16h load_b_frag_lds(const _Float16* lds, int nn, int lane) {
  const int n_local = (nn << 4) + (lane & 15);
  const int kb = (lane >> 4) << 4;
  const _Float16* p = lds + n_local * 32 + kb;
  v16h b;
#pragma unroll
  for (int e = 0; e < 16; ++e) b[e] = p[e];
  return b;
}

// C/D 16x16 f32: VGPR j -> row m0 + (lane>=16 ? 8:0) + j, col n0 + (lane&15)
__device__ __forceinline__ void store_c_frag(float* C, int ldc, int m0, int n0, int lane,
                                             v8f acc, const float* bias) {
  const int c  = n0 + (lane & 15);
  const int rb = m0 + ((lane >> 4) << 3);
  const float bv = bias[c];
#pragma unroll
  for (int j = 0; j < 8; ++j) C[(size_t)(rb + j) * ldc + c] = acc[j] + bv;
}

#if USE_ASYNC_LDS
// Exact pointee type from the builtin's signature:
//   '__attribute__((__vector_size__(4 * sizeof(int)))) int __device__ *'
typedef int v4i_vs __attribute__((vector_size(4 * sizeof(int))));
typedef __attribute__((address_space(1))) v4i_vs* as1_v4i_p;
typedef __attribute__((address_space(3))) v4i_vs* as3_v4i_p;

__device__ __forceinline__ void async_b128(const void* g, void* l) {
  __builtin_amdgcn_global_load_async_to_lds_b128(
      (as1_v4i_p)(v4i_vs*)g,  // strip const, then addrspacecast generic -> global
      (as3_v4i_p)(v4i_vs*)l,  // addrspacecast generic -> LDS
      0, 0);
}

// Stage one 64(N) x 32(K) f16 weight tile into LDS: 4 KB, 8 async b128 ops
// (each op: 32 lanes x 16B). Tracked by ASYNCcnt.
__device__ __forceinline__ void async_copy_btile(const _Float16* W, int ldw, int n0, int k0,
                                                 _Float16* lds, int lane) {
  const int rsub = lane >> 2; // 0..7
  const int seg  = lane & 3;  // 0..3, 16B segments of a 64B row
#pragma unroll
  for (int i = 0; i < 8; ++i) {
    const int row = (i << 3) + rsub;
    const _Float16* g = W + (size_t)(n0 + row) * ldw + k0 + (seg << 3);
    _Float16* l = lds + row * 32 + (seg << 3);
    async_b128(g, l);
  }
}

__device__ __forceinline__ void wait_async_le8() {
#if __has_builtin(__builtin_amdgcn_s_wait_asynccnt)
  __builtin_amdgcn_s_wait_asynccnt(8);
#else
  asm volatile("s_wait_asynccnt 0x8" ::: "memory");
#endif
}
__device__ __forceinline__ void wait_async_0() {
#if __has_builtin(__builtin_amdgcn_s_wait_asynccnt)
  __builtin_amdgcn_s_wait_asynccnt(0);
#else
  asm volatile("s_wait_asynccnt 0x0" ::: "memory");
#endif
}
#endif // USE_ASYNC_LDS

// ---- Generic WMMA GEMM: one wave per 32(M) x 64(N) block, 2x4 tiles ----------
// grid.x = N/64, grid.y = M/32, grid.z selects job (dual-problem launches).
__global__ void gemm_wmma(GemmJob j0, GemmJob j1, int lda, int ldw, int ldc, int K) {
  const GemmJob jb = (blockIdx.z == 0) ? j0 : j1;
  const int lane = threadIdx.x;
  const int n0 = blockIdx.x * 64;
  const int m0 = blockIdx.y * 32;
  v8f acc[2][4] = {};
#if USE_ASYNC_LDS
  __shared__ _Float16 lbuf[2][64 * 32]; // double-buffered weight tile (2 x 4 KB)
  async_copy_btile(jb.W, ldw, n0, 0, lbuf[0], lane);
  int cur = 0;
  for (int k0 = 0; k0 < K; k0 += 32) {
    if (k0 + 32 < K) {
      async_copy_btile(jb.W, ldw, n0, k0 + 32, lbuf[cur ^ 1], lane);
      wait_async_le8(); // previous tile (async loads complete in order) is in LDS
    } else {
      wait_async_0();
    }
    v16h af0 = load_a_frag(jb.A, lda, m0,      k0, lane);
    v16h af1 = load_a_frag(jb.A, lda, m0 + 16, k0, lane);
#pragma unroll
    for (int nn = 0; nn < 4; ++nn) {
      v16h bf = load_b_frag_lds(lbuf[cur], nn, lane);
      acc[0][nn] = __builtin_amdgcn_wmma_f32_16x16x32_f16(false, af0, false, bf, (short)0, acc[0][nn], false, false);
      acc[1][nn] = __builtin_amdgcn_wmma_f32_16x16x32_f16(false, af1, false, bf, (short)0, acc[1][nn], false, false);
    }
    cur ^= 1;
  }
#else
  for (int k0 = 0; k0 < K; k0 += 32) {
    if (k0 + 32 < K) { // pull next weight slab toward the caches
      __builtin_prefetch(jb.W + (size_t)(n0 + (lane & 15)) * ldw + (k0 + 32), 0, 3);
      __builtin_prefetch(jb.W + (size_t)(n0 + 32 + (lane & 15)) * ldw + (k0 + 32), 0, 3);
    }
    v16h af0 = load_a_frag(jb.A, lda, m0,      k0, lane);
    v16h af1 = load_a_frag(jb.A, lda, m0 + 16, k0, lane);
#pragma unroll
    for (int nn = 0; nn < 4; ++nn) {
      v16h bf = load_b_frag_g(jb.W, ldw, n0 + nn * 16, k0, lane);
      acc[0][nn] = __builtin_amdgcn_wmma_f32_16x16x32_f16(false, af0, false, bf, (short)0, acc[0][nn], false, false);
      acc[1][nn] = __builtin_amdgcn_wmma_f32_16x16x32_f16(false, af1, false, bf, (short)0, acc[1][nn], false, false);
    }
  }
#endif
#pragma unroll
  for (int nn = 0; nn < 4; ++nn) {
    store_c_frag(jb.C, ldc, m0,      n0 + nn * 16, lane, acc[0][nn], jb.bias);
    store_c_frag(jb.C, ldc, m0 + 16, n0 + nn * 16, lane, acc[1][nn], jb.bias);
  }
}

// ---- Elementwise / small kernels --------------------------------------------

__device__ __forceinline__ float sigm(float x) { return 1.0f / (1.0f + __expf(-x)); }

__global__ void cvt_f32_f16(const float* __restrict__ s, _Float16* __restrict__ d, int n) {
  int i = blockIdx.x * blockDim.x + threadIdx.x;
  if (i < n) d[i] = (_Float16)s[i];
}

__global__ void embed_gather(const int* __restrict__ motion, const float* __restrict__ emb,
                             _Float16* __restrict__ x) {
  int idx = blockIdx.x * blockDim.x + threadIdx.x; // B*S*E
  if (idx >= B_ * S_ * E_) return;
  int bs = idx >> 9;        // /E_
  int e  = idx & (E_ - 1);
  x[idx] = (_Float16)emb[(size_t)motion[bs] * E_ + e];
}

// GRU gate fusion (encoder). grid.z = direction. Writes h (f32+f16) and y[:,t,dir*H:+H] (f16).
__global__ void enc_gru_elem(DirPtrs d0, DirPtrs d1, _Float16* __restrict__ y, int s) {
  const int dir = blockIdx.z;
  const DirPtrs d = dir ? d1 : d0;
  const int t = dir ? (S_ - 1 - s) : s;
  const int idx = blockIdx.x * blockDim.x + threadIdx.x;
  if (idx >= B_ * H_) return;
  const int b = idx >> 10, j = idx & (H_ - 1);
  const float* xr = d.xg + ((size_t)b * S_ + t) * G_;
  const float* gr = d.gh + (size_t)b * G_;
  float r = sigm(xr[j] + gr[j]);
  float z = sigm(xr[H_ + j] + gr[H_ + j]);
  float n = tanhf(xr[2 * H_ + j] + r * gr[2 * H_ + j]);
  float hn = (1.0f - z) * n + z * d.h[idx];
  d.h[idx]  = hn;
  d.hf[idx] = (_Float16)hn;
  y[((size_t)b * S_ + t) * (2 * H_) + (size_t)dir * H_ + j] = (_Float16)hn;
}

// GRU gate fusion (decoder). gates = [gi | gh], each [B,3H]. Writes h (f32+f16) and dec[:,t,:] (f16).
__global__ void dec_gru_elem(const float* __restrict__ gates, const float* __restrict__ h_in,
                             float* __restrict__ h_out, _Float16* __restrict__ hf_out,
                             _Float16* __restrict__ dec_t) {
  const int idx = blockIdx.x * blockDim.x + threadIdx.x;
  if (idx >= B_ * H_) return;
  const int b = idx >> 10, j = idx & (H_ - 1);
  const float* gi = gates + (size_t)b * G_;
  const float* gh = gates + (size_t)B_ * G_ + (size_t)b * G_;
  float r = sigm(gi[j] + gh[j]);
  float z = sigm(gi[H_ + j] + gh[H_ + j]);
  float n = tanhf(gi[2 * H_ + j] + r * gh[2 * H_ + j]);
  float hn = (1.0f - z) * n + z * h_in[idx];
  h_out[idx]  = hn;
  hf_out[idx] = (_Float16)hn;
  dec_t[(size_t)b * (T_ * H_) + j] = (_Float16)hn;
}

// attn_logits[b,s] = dot(concat(dec_emb[0], enc_hidden[b]), attn_w[s]) + attn_b[s]
__global__ void attn_logits_kernel(const float* __restrict__ dec_emb0,
                                   const float* __restrict__ enc_hidden,
                                   const float* __restrict__ attn_w,
                                   const float* __restrict__ attn_b,
                                   float* __restrict__ out) {
  int idx = blockIdx.x * blockDim.x + threadIdx.x; // B*S
  if (idx >= B_ * S_) return;
  int b = idx >> 6, s = idx & (S_ - 1);
  const float* w = attn_w + (size_t)s * (2 * H_);
  float acc = attn_b[s];
  for (int i = 0; i < H_; ++i) acc += dec_emb0[i] * w[i];
  const float* hb = enc_hidden + (size_t)b * H_;
  for (int i = 0; i < H_; ++i) acc += hb[i] * w[H_ + i];
  out[idx] = acc;
}

__global__ void softmax_rows_small(float* __restrict__ x) { // [B,S] in place, S=64
  int b = threadIdx.x;
  if (b >= B_) return;
  float* row = x + (size_t)b * S_;
  float m = -3.0e38f;
  for (int s = 0; s < S_; ++s) m = fmaxf(m, row[s]);
  float sum = 0.0f;
  for (int s = 0; s < S_; ++s) sum += __expf(row[s] - m);
  float inv = 1.0f / sum;
  for (int s = 0; s < S_; ++s) row[s] = __expf(row[s] - m) * inv;
}

// applied[b,j] = sum_s aw[b,s] * enc_out[b,s,j]   (enc_out stored f16)
__global__ void attn_apply(const float* __restrict__ aw, const _Float16* __restrict__ enc_out,
                           float* __restrict__ applied) {
  int idx = blockIdx.x * blockDim.x + threadIdx.x; // B*2H
  if (idx >= B_ * 2 * H_) return;
  int b = idx >> 11, j = idx & (2 * H_ - 1);
  float acc = 0.0f;
  for (int s = 0; s < S_; ++s)
    acc += aw[b * S_ + s] * (float)enc_out[((size_t)b * S_ + s) * (2 * H_) + j];
  applied[idx] = acc;
}

// cat[b,:] = [dec_emb[0] (H) | applied[b] (2H)]  (f16, GEMM A operand)
__global__ void build_cat(const float* __restrict__ dec_emb0, const float* __restrict__ applied,
                          _Float16* __restrict__ cat) {
  int idx = blockIdx.x * blockDim.x + threadIdx.x; // B*3H
  if (idx >= B_ * G_) return;
  int b = idx / G_, j = idx - b * G_;
  cat[idx] = (_Float16)((j < H_) ? dec_emb0[j] : applied[(size_t)b * (2 * H_) + (j - H_)]);
}

// In-place log_softmax over rows of length V, one workgroup per row.
__global__ void log_softmax_rows(float* __restrict__ x) {
  __shared__ float red[256];
  float* row = x + (size_t)blockIdx.x * V_;
  const int tid = threadIdx.x;
  float m = -3.0e38f;
  for (int i = tid; i < V_; i += 256) m = fmaxf(m, row[i]);
  red[tid] = m; __syncthreads();
  for (int s = 128; s > 0; s >>= 1) {
    if (tid < s) red[tid] = fmaxf(red[tid], red[tid + s]);
    __syncthreads();
  }
  m = red[0]; __syncthreads();
  float sum = 0.0f;
  for (int i = tid; i < V_; i += 256) sum += __expf(row[i] - m);
  red[tid] = sum; __syncthreads();
  for (int s = 128; s > 0; s >>= 1) {
    if (tid < s) red[tid] += red[tid + s];
    __syncthreads();
  }
  const float L = m + __logf(red[0]);
  for (int i = tid; i < V_; i += 256) row[i] -= L;
}

// ---- Host-side orchestration -------------------------------------------------

extern "C" void kernel_launch(void* const* d_in, const int* in_sizes, int n_in,
                              void* d_out, int out_size, void* d_ws, size_t ws_size,
                              hipStream_t stream) {
  (void)in_sizes; (void)n_in; (void)out_size; (void)ws_size;
  // Input leaf order (pytree-flattened setup_inputs dict):
  const int*   motion  = (const int*)  d_in[0];
  const float* enc_emb = (const float*)d_in[1];
  const float* dec_emb = (const float*)d_in[2];
  const float *g_wih[2][2], *g_whh[2][2], *g_bih[2][2], *g_bhh[2][2];
  for (int l = 0; l < 2; ++l)
    for (int d = 0; d < 2; ++d) {
      int base = 3 + 4 * (2 * l + d);
      g_wih[l][d] = (const float*)d_in[base + 0];
      g_whh[l][d] = (const float*)d_in[base + 1];
      g_bih[l][d] = (const float*)d_in[base + 2];
      g_bhh[l][d] = (const float*)d_in[base + 3];
    }
  const float* dec_wih = (const float*)d_in[19];
  const float* dec_whh = (const float*)d_in[20];
  const float* dec_bih = (const float*)d_in[21];
  const float* dec_bhh = (const float*)d_in[22];
  const float* attn_w  = (const float*)d_in[23];
  const float* attn_b  = (const float*)d_in[24];
  const float* comb_w  = (const float*)d_in[25];
  const float* comb_b  = (const float*)d_in[26];
  const float* out_w   = (const float*)d_in[27];
  const float* out_b   = (const float*)d_in[28];

  // Workspace bump allocator (~230 MB total)
  char* base = (char*)d_ws;
  size_t off = 0;
  auto alloc = [&](size_t bytes) -> void* {
    off = (off + 255) & ~(size_t)255;
    void* p = base + off;
    off += bytes;
    return p;
  };

  // f16 weight copies (L2-resident working sets)
  _Float16* wf_wih[2][2]; _Float16* wf_whh[2][2];
  const int din_l[2] = { E_, 2 * H_ };
  for (int l = 0; l < 2; ++l)
    for (int d = 0; d < 2; ++d) {
      wf_wih[l][d] = (_Float16*)alloc((size_t)G_ * din_l[l] * 2);
      wf_whh[l][d] = (_Float16*)alloc((size_t)G_ * H_ * 2);
    }
  _Float16* wf_dec_wih = (_Float16*)alloc((size_t)G_ * H_ * 2);
  _Float16* wf_dec_whh = (_Float16*)alloc((size_t)G_ * H_ * 2);
  _Float16* wf_comb    = (_Float16*)alloc((size_t)H_ * G_ * 2);
  _Float16* wf_out     = (_Float16*)alloc((size_t)V_ * H_ * 2);

  // Activations: GEMM A-operands in f16, GEMM C outputs / GRU state in f32
  _Float16* xbuf      = (_Float16*)alloc((size_t)B_ * S_ * E_ * 2);
  float*    xg0       = (float*)   alloc((size_t)B_ * S_ * G_ * 4);
  float*    xg1       = (float*)   alloc((size_t)B_ * S_ * G_ * 4);
  _Float16* y0h       = (_Float16*)alloc((size_t)B_ * S_ * 2 * H_ * 2);
  _Float16* enc_out_h = (_Float16*)alloc((size_t)B_ * S_ * 2 * H_ * 2);
  float*    gh2       = (float*)   alloc((size_t)2 * B_ * G_ * 4);
  float*    henc      = (float*)   alloc((size_t)2 * B_ * H_ * 4);
  _Float16* henc_h    = (_Float16*)alloc((size_t)2 * B_ * H_ * 2);
  float*    attnlog   = (float*)   alloc((size_t)B_ * S_ * 4);
  float*    applied   = (float*)   alloc((size_t)B_ * 2 * H_ * 4);
  _Float16* cath      = (_Float16*)alloc((size_t)B_ * G_ * 2);
  float*    out0      = (float*)   alloc((size_t)B_ * H_ * 4);
  _Float16* out0h     = (_Float16*)alloc((size_t)B_ * H_ * 2);
  float*    dgates    = (float*)   alloc((size_t)2 * B_ * G_ * 4);
  float*    hdec      = (float*)   alloc((size_t)B_ * H_ * 4);
  _Float16* hdec_h    = (_Float16*)alloc((size_t)B_ * H_ * 2);
  _Float16* dec_h     = (_Float16*)alloc((size_t)B_ * T_ * H_ * 2);

  auto cvt = [&](const float* s, _Float16* d, size_t n) {
    cvt_f32_f16<<<dim3((unsigned)((n + 255) / 256)), 256, 0, stream>>>(s, d, (int)n);
  };
  for (int l = 0; l < 2; ++l)
    for (int d = 0; d < 2; ++d) {
      cvt(g_wih[l][d], wf_wih[l][d], (size_t)G_ * din_l[l]);
      cvt(g_whh[l][d], wf_whh[l][d], (size_t)G_ * H_);
    }
  cvt(dec_wih, wf_dec_wih, (size_t)G_ * H_);
  cvt(dec_whh, wf_dec_whh, (size_t)G_ * H_);
  cvt(comb_w,  wf_comb,    (size_t)H_ * G_);
  cvt(out_w,   wf_out,     (size_t)V_ * H_);

  // 1) Embedding gather (writes f16 GEMM operand)
  embed_gather<<<dim3((B_ * S_ * E_ + 255) / 256), 256, 0, stream>>>(motion, enc_emb, xbuf);

  // 2) Encoder: 2 layers x bidirectional GRU
  const dim3 elemGrid((B_ * H_ + 255) / 256, 1, 2);
  for (int l = 0; l < 2; ++l) {
    const _Float16* lin = (l == 0) ? xbuf : y0h;
    _Float16* yout = (l == 0) ? y0h : enc_out_h;
    const int Din = din_l[l];
    (void)hipMemsetAsync(henc,   0, (size_t)2 * B_ * H_ * 4, stream); // h0 = 0 (f32)
    (void)hipMemsetAsync(henc_h, 0, (size_t)2 * B_ * H_ * 2, stream); // h0 = 0 (f16)
    // Precompute input gates for both directions in one launch (grid.z=2)
    GemmJob xj0{lin, wf_wih[l][0], g_bih[l][0], xg0};
    GemmJob xj1{lin, wf_wih[l][1], g_bih[l][1], xg1};
    gemm_wmma<<<dim3(G_ / 64, (B_ * S_) / 32, 2), 32, 0, stream>>>(xj0, xj1, Din, Din, G_, Din);
    // 64 sequential steps; forward + backward run in parallel via grid.z
    for (int s = 0; s < S_; ++s) {
      GemmJob hj0{henc_h,            wf_whh[l][0], g_bhh[l][0], gh2};
      GemmJob hj1{henc_h + B_ * H_,  wf_whh[l][1], g_bhh[l][1], gh2 + (size_t)B_ * G_};
      gemm_wmma<<<dim3(G_ / 64, 1, 2), 32, 0, stream>>>(hj0, hj1, H_, H_, G_, H_);
      DirPtrs p0{xg0, gh2,                   henc,            henc_h};
      DirPtrs p1{xg1, gh2 + (size_t)B_ * G_, henc + B_ * H_,  henc_h + B_ * H_};
      enc_gru_elem<<<elemGrid, 256, 0, stream>>>(p0, p1, yout, s);
    }
  }
  float*    enc_hidden   = henc   + B_ * H_; // layer-1 backward final hidden (f32)
  _Float16* enc_hidden_h = henc_h + B_ * H_; // and its f16 mirror

  // 3) Attention for the first decoder input
  attn_logits_kernel<<<dim3((B_ * S_ + 255) / 256), 256, 0, stream>>>(dec_emb, enc_hidden,
                                                                      attn_w, attn_b, attnlog);
  softmax_rows_small<<<1, 32, 0, stream>>>(attnlog);
  attn_apply<<<dim3((B_ * 2 * H_ + 255) / 256), 256, 0, stream>>>(attnlog, enc_out_h, applied);
  build_cat<<<dim3((B_ * G_ + 255) / 256), 256, 0, stream>>>(dec_emb, applied, cath);
  GemmJob cj{cath, wf_comb, comb_b, out0};
  gemm_wmma<<<dim3(H_ / 64, 1, 1), 32, 0, stream>>>(cj, cj, G_, G_, H_, G_);
  cvt(out0, out0h, (size_t)B_ * H_);

  // 4) Decoder: 128 sequential GRU-cell steps; gi and gh GEMMs share a launch
  const dim3 delemGrid((B_ * H_ + 255) / 256, 1, 1);
  {
    GemmJob dj0{out0h,        wf_dec_wih, dec_bih, dgates};
    GemmJob dj1{enc_hidden_h, wf_dec_whh, dec_bhh, dgates + (size_t)B_ * G_};
    gemm_wmma<<<dim3(G_ / 64, 1, 2), 32, 0, stream>>>(dj0, dj1, H_, H_, G_, H_);
    dec_gru_elem<<<delemGrid, 256, 0, stream>>>(dgates, enc_hidden, hdec, hdec_h, dec_h + 0 * H_);
  }
  for (int t = 1; t < T_; ++t) {
    GemmJob dj0{hdec_h, wf_dec_wih, dec_bih, dgates};
    GemmJob dj1{hdec_h, wf_dec_whh, dec_bhh, dgates + (size_t)B_ * G_};
    gemm_wmma<<<dim3(G_ / 64, 1, 2), 32, 0, stream>>>(dj0, dj1, H_, H_, G_, H_);
    dec_gru_elem<<<delemGrid, 256, 0, stream>>>(dgates, hdec, hdec, hdec_h, dec_h + (size_t)t * H_);
  }

  // 5) Output projection into d_out (268 GFLOP; out_w f16 = 65 MB, L2-resident)
  GemmJob pj{dec_h, wf_out, out_b, (float*)d_out};
  gemm_wmma<<<dim3(V_ / 64, (B_ * T_) / 32, 1), 32, 0, stream>>>(pj, pj, H_, H_, V_, H_);

  // 6) In-place log-softmax over V per (b,t) row
  log_softmax_rows<<<dim3(B_ * T_), 256, 0, stream>>>((float*)d_out);
}
